// DNASPreBasicBlockMemory_91250875171258
// MI455X (gfx1250) — matrix operand order repaired and checked by reading, not compiled, not run on hardware
//
#include <hip/hip_runtime.h>

// ---------------------------------------------------------------------------
// DNAS pre-act block for MI455X (gfx1250, wave32, WMMA).
// 9 candidates folded: activations quantized once per ba (3 tensors, f16,
// NHWC zero-padded), bw candidates folded into effective weights, and the
// 3 remaining convs fused into ONE implicit GEMM per layer:
//   M=128 (C_out), N=100352 (32*56*56 pixels), K=3456 = [ba][kh][kw][ci]
// ci innermost => im2col K-runs are contiguous in memory (vector/async loads).
// f16 inputs / f32 accumulate via v_wmma_f32_16x16x32_f16, double-buffered LDS.
// ---------------------------------------------------------------------------

typedef _Float16 v16h __attribute__((ext_vector_type(16)));
typedef _Float16 v8h  __attribute__((ext_vector_type(8)));
typedef float    v8f  __attribute__((ext_vector_type(8)));
typedef int      v4i  __attribute__((ext_vector_type(4)));

constexpr int NB   = 32;            // batch
constexpr int C    = 128;           // channels
constexpr int H    = 56;
constexpr int W    = 56;
constexpr int HW   = H * W;         // 3136
constexpr int NPIX = NB * HW;       // 100352  (GEMM N)
constexpr int PADH = 58;
constexpr int PADW = 58;
constexpr long long PAD_ELEMS = (long long)NB * PADH * PADW * C; // 13,778,944
constexpr int KCAND = 9 * C;        // 1152 ( [kh][kw][ci] )
constexpr int KTOT  = 3 * KCAND;    // 3456 (3 ba-candidates fused)
constexpr int WCAND = C * C * 9;    // 147456 (one candidate weight tensor)

#if defined(__has_builtin)
#if __has_builtin(__builtin_amdgcn_global_load_async_to_lds_b128)
#define HAVE_ASYNC_LDS 1
#endif
#endif

__device__ __forceinline__ void wait_async_all() {
#if defined(__has_builtin) && __has_builtin(__builtin_amdgcn_s_wait_asynccnt)
  __builtin_amdgcn_s_wait_asynccnt(0);
#else
  asm volatile("s_wait_asynccnt 0x0" ::: "memory");
#endif
}

// ---------------- tiny: gumbel-softmax probabilities (both layers) ----------
__global__ void k_probs(const float* p1, const float* gn1,
                        const float* p2, const float* gn2,
                        const float* tau, float* probs) {
  if (threadIdx.x == 0) {
    float t = tau[0];
    for (int L = 0; L < 2; ++L) {
      const float* p = L ? p2 : p1;
      const float* g = L ? gn2 : gn1;
      float z[9], m = -1e30f, s = 0.f;
      for (int k = 0; k < 9; ++k) { z[k] = (p[k] + g[k]) / t; m = fmaxf(m, z[k]); }
      for (int k = 0; k < 9; ++k) { z[k] = expf(z[k] - m); s += z[k]; }
      for (int k = 0; k < 9; ++k) probs[L * 9 + k] = z[k] / s;
    }
  }
}

// ---------------- per-candidate max|tanh(w)| reduction ----------------------
__global__ void k_wmax(const float* w1, const float* w2, float* wmax) {
  const int t = blockIdx.x;                         // 0..17
  const float* base = (t < 9) ? (w1 + (size_t)t * WCAND)
                              : (w2 + (size_t)(t - 9) * WCAND);
  float m = 0.f;
  for (int i = threadIdx.x; i < WCAND; i += 256)
    m = fmaxf(m, fabsf(tanhf(base[i])));
  __shared__ float red[256];
  red[threadIdx.x] = m;
  __syncthreads();
  for (int s = 128; s > 0; s >>= 1) {
    if (threadIdx.x < s) red[threadIdx.x] = fmaxf(red[threadIdx.x], red[threadIdx.x + s]);
    __syncthreads();
  }
  if (threadIdx.x == 0) wmax[t] = red[0];
}

// ---------------- fold 3 bw-candidates into one f16 W_eff per ba ------------
// weff[L][co][kt], kt = ba*1152 + (kh*3+kw)*128 + ci  (ci innermost, matches
// NHWC activation layout so B-side im2col K-runs are contiguous).
__global__ void k_weff(const float* w1, const float* w2,
                       const float* probs, const float* wmax, _Float16* weff) {
  const int idx = blockIdx.x * 256 + threadIdx.x;   // < 2*128*3456 = 884736
  const int L   = idx / (C * KTOT);
  const int rem = idx % (C * KTOT);
  const int co  = rem / KTOT;
  const int kt  = rem % KTOT;
  const int ba  = kt / KCAND;
  const int t2  = kt % KCAND;
  const int r9  = t2 >> 7;                          // kh*3+kw
  const int ci  = t2 & 127;
  const float* wsrc = L ? w2 : w1;
  const float nq[3] = {3.f, 15.f, 255.f};           // 2^{2,4,8}-1
  float acc = 0.f;
  for (int bw = 0; bw < 3; ++bw) {
    const int cand = bw * 3 + ba;                   // candidate order: bw major
    const float w  = wsrc[(size_t)cand * WCAND + (size_t)co * 1152 + ci * 9 + r9];
    const float wt = tanhf(w);
    const float wn = wt / (2.f * wmax[L * 9 + cand]) + 0.5f;
    const float q  = rintf(wn * nq[bw]) / nq[bw];   // RNE matches jnp.round
    acc += probs[L * 9 + cand] * (2.f * q - 1.f);
  }
  weff[idx] = (_Float16)acc;
}

// ---------------- BN (training-mode batch stats) -> scale/shift -------------
__global__ void k_bnstats(const float* __restrict__ x, const float* gamma,
                          const float* beta, float* scale, float* shift) {
  const int c = blockIdx.x;                         // one block per channel
  float s = 0.f, s2 = 0.f;
  for (int i = threadIdx.x; i < NB * HW; i += 256) {
    const int n = i / HW, r = i % HW;
    const float v = x[((size_t)n * C + c) * HW + r];
    s += v; s2 += v * v;
  }
  __shared__ float rs[256], rs2[256];
  rs[threadIdx.x] = s; rs2[threadIdx.x] = s2;
  __syncthreads();
  for (int k = 128; k > 0; k >>= 1) {
    if (threadIdx.x < k) { rs[threadIdx.x] += rs[threadIdx.x + k]; rs2[threadIdx.x] += rs2[threadIdx.x + k]; }
    __syncthreads();
  }
  if (threadIdx.x == 0) {
    const float cnt = (float)(NB * HW);
    const float mu  = rs[0] / cnt;
    const float var = rs2[0] / cnt - mu * mu;       // biased var (jnp.var)
    const float rsg = rsqrtf(var + 1e-5f);
    const float sc  = gamma[c] * rsg;
    scale[c] = sc;
    shift[c] = beta[c] - mu * sc;
  }
}

// ------- fused BN + ReLU + clip + 3-way quantize -> padded NHWC f16 ---------
// gid ordered [n][oh][ow][c] (c fastest) => fully coalesced f16 stores.
__global__ void k_quant(const float* __restrict__ in, const float* __restrict__ scale,
                        const float* __restrict__ shift, _Float16* __restrict__ hq) {
  const size_t gid = (size_t)blockIdx.x * 256 + threadIdx.x;   // < 12,845,056
  const int c   = (int)(gid & (C - 1));
  const size_t pix = gid >> 7;                      // /C
  const size_t n = pix / HW;
  const int r   = (int)(pix % HW);
  const int oh = r / W, ow = r % W;
  float y = in[(n * C + c) * HW + r] * scale[c] + shift[c];
  y = fminf(fmaxf(y, 0.f), 1.f);                    // relu + clip[0,1]
  const size_t pa = ((n * PADH + (oh + 1)) * PADW + (ow + 1)) * C + c;
  hq[pa]                         = (_Float16)(rintf(y * 3.f)   * (1.f / 3.f));
  hq[(size_t)PAD_ELEMS     + pa] = (_Float16)(rintf(y * 15.f)  * (1.f / 15.f));
  hq[(size_t)PAD_ELEMS * 2 + pa] = (_Float16)(rintf(y * 255.f) * (1.f / 255.f));
}

// ---------------- implicit-GEMM conv, WMMA f32_16x16x32_f16 -----------------
// Block = 256 threads (8 wave32 waves). Block tile: M=128 x N=64, K-step 64.
// Double-buffered LDS B-tiles (ping-pong, ONE barrier per K-step); staging is
// one contiguous 16B transfer per thread per 32-chunk (async-to-LDS if the
// builtin exists, else b128 load + ds_store_b128).
__device__ __forceinline__ void stage_b(const _Float16* __restrict__ hq,
                                        size_t colpix, int k, _Float16* dst) {
  const int ba = k / KCAND;
  const int t2 = k - ba * KCAND;
  const int r9 = t2 >> 7;                           // kh*3+kw (fixed per chunk)
  const int ci = t2 & 127;
  const int kh = r9 / 3, kw = r9 - kh * 3;
  const _Float16* src = hq + (size_t)ba * PAD_ELEMS + colpix +
                        ((size_t)kh * PADW + kw) * C + ci;
#if defined(HAVE_ASYNC_LDS)
  // builtin signature (from clang diagnostic): arg0 = global int4* (AS1),
  // arg1 = LDS int4* (AS3), then imm offset + cpol.
  __builtin_amdgcn_global_load_async_to_lds_b128(
      (__attribute__((address_space(1))) v4i*)(void*)src,
      (__attribute__((address_space(3))) v4i*)(void*)dst, 0, 0);
#else
  *(v8h*)dst = *(const v8h*)src;
#endif
}

template <bool HASRES>
__global__ __launch_bounds__(256) void k_conv(const _Float16* __restrict__ hq,
                                              const _Float16* __restrict__ weff,
                                              const float* __restrict__ residual,
                                              float* __restrict__ out) {
  __shared__ _Float16 Btile[2][64 * 64];            // [buf][col][k], 16 KB
  const int t    = threadIdx.x;
  const int lane = t & 31;
  const int wv   = t >> 5;
  const int pb   = blockIdx.x * 64;                 // pixel-tile base (N dim)

  // Loader role: col = t/4, two 8-half sub-chunks at lk0 and lk0+32.
  const int lcol = t >> 2;
  const int lk0  = (t & 3) * 8;
  const int p  = pb + lcol;
  const int n  = p / HW;
  const int rr = p % HW;
  const int oh = rr / W, ow = rr % W;
  const size_t colpix = ((size_t)(n * PADH + oh) * PADW + ow) * C;

  // A fragment addressing per ISA: lanes 0-15 -> K {0..7,16..23}, lanes 16-31
  // -> K {8..15,24..31}; M = lane % 16 within the wave's 16-row strip.
  const int m    = (wv << 4) + (lane & 15);
  const int ksel = (lane < 16) ? 0 : 8;
  const int bk   = (lane < 16) ? 0 : 8;
  const int bn   = lane & 15;
  const _Float16* arow = weff + (size_t)m * KTOT;

  v8f acc[4] = {};

  constexpr int NIT = KTOT / 64;                    // 54
  // prologue: stage K-chunk 0 into buffer 0
  stage_b(hq, colpix, lk0,      &Btile[0][lcol * 64 + lk0]);
  stage_b(hq, colpix, lk0 + 32, &Btile[0][lcol * 64 + lk0 + 32]);

  for (int it = 0; it < NIT; ++it) {
    const int cur = it & 1;
#if defined(HAVE_ASYNC_LDS)
    wait_async_all();                               // own async loads landed
#endif
    __syncthreads();                                // all waves' data visible
    if (it + 1 < NIT) {
      const int kn = (it + 1) * 64;
      stage_b(hq, colpix, kn + lk0,      &Btile[cur ^ 1][lcol * 64 + lk0]);
      stage_b(hq, colpix, kn + lk0 + 32, &Btile[cur ^ 1][lcol * 64 + lk0 + 32]);
    }
    __builtin_prefetch(arow + (it + 2) * 64, 0, 0); // weight stream prefetch

#pragma unroll
    for (int c2 = 0; c2 < 2; ++c2) {
      const int kt = it * 64 + c2 * 32;
      const v8h alo = *(const v8h*)(arow + kt + ksel);
      const v8h ahi = *(const v8h*)(arow + kt + ksel + 16);
      v16h a;
#pragma unroll
      for (int i = 0; i < 8; ++i) { a[i] = alo[i]; a[8 + i] = ahi[i]; }
#pragma unroll
      for (int s = 0; s < 4; ++s) {
        const _Float16* bp = &Btile[cur][(s * 16 + bn) * 64 + c2 * 32 + bk];
        const v8h blo = *(const v8h*)(bp);
        const v8h bhi = *(const v8h*)(bp + 16);
        v16h b;
#pragma unroll
        for (int i = 0; i < 8; ++i) { b[i] = blo[i]; b[8 + i] = bhi[i]; }
        acc[s] = __builtin_amdgcn_wmma_f32_16x16x32_f16(
            false, a, false, b, (short)0, acc[s], false, false);
      }
    }
    // no trailing barrier needed: next overwrite of `cur` happens after the
    // next iteration's top barrier, which follows every wave's reads above.
  }

  // ---- epilogue: D layout lanes 0-15 -> M=v, lanes 16-31 -> M=8+v ----
  const int mo = (lane >= 16) ? 8 : 0;
  const int nn = lane & 15;
#pragma unroll
  for (int s = 0; s < 4; ++s) {
    const int pp = pb + s * 16 + nn;
    const int ni = pp / HW;
    const int sr = pp % HW;
#pragma unroll
    for (int v = 0; v < 8; ++v) {
      const int co = (wv << 4) + mo + v;
      const size_t oidx = ((size_t)ni * C + co) * HW + sr;
      float val = acc[s][v];
      if constexpr (HASRES) val += residual[oidx];
      out[oidx] = val;
    }
  }
}

// ---------------------------------------------------------------------------
extern "C" void kernel_launch(void* const* d_in, const int* in_sizes, int n_in,
                              void* d_out, int out_size, void* d_ws, size_t ws_size,
                              hipStream_t stream) {
  (void)in_sizes; (void)n_in; (void)out_size; (void)ws_size;
  const float* x       = (const float*)d_in[0];
  const float* gamma1  = (const float*)d_in[1];
  const float* beta1   = (const float*)d_in[2];
  const float* gamma2  = (const float*)d_in[3];
  const float* beta2   = (const float*)d_in[4];
  const float* conv1_w = (const float*)d_in[5];
  const float* conv2_w = (const float*)d_in[6];
  const float* p1      = (const float*)d_in[7];
  const float* p2      = (const float*)d_in[8];
  const float* gn1     = (const float*)d_in[9];
  const float* gn2     = (const float*)d_in[10];
  const float* tau     = (const float*)d_in[11];
  float* out = (float*)d_out;

  char* ws = (char*)d_ws;
  float*    probs  = (float*)(ws + 0);                 // 18 f
  float*    wmax   = (float*)(ws + 256);               // 18 f
  float*    scale1 = (float*)(ws + 1024);              // 128 f each
  float*    shift1 = (float*)(ws + 1536);
  float*    scale2 = (float*)(ws + 2048);
  float*    shift2 = (float*)(ws + 2560);
  _Float16* weff   = (_Float16*)(ws + 4096);           // 2*128*3456 halves
  _Float16* hq     = (_Float16*)(ws + (size_t)2097152);     // 3 padded f16 acts
  float*    c1     = (float*)(ws + (size_t)84770816);       // layer-1 conv out

  // weight preprocessing (tiny vs. conv cost)
  k_probs<<<1, 32, 0, stream>>>(p1, gn1, p2, gn2, tau, probs);
  k_wmax<<<18, 256, 0, stream>>>(conv1_w, conv2_w, wmax);
  k_weff<<<3456, 256, 0, stream>>>(conv1_w, conv2_w, probs, wmax, weff);

  // zero padded activation borders (interior overwritten by k_quant)
  (void)hipMemsetAsync(hq, 0, (size_t)3 * PAD_ELEMS * sizeof(_Float16), stream);

  // layer 1: BN stats -> quantize -> fused implicit GEMM
  k_bnstats<<<C, 256, 0, stream>>>(x, gamma1, beta1, scale1, shift1);
  k_quant<<<NPIX * C / 256, 256, 0, stream>>>(x, scale1, shift1, hq);
  k_conv<false><<<NPIX / 64, 256, 0, stream>>>(hq, weff, nullptr, c1);

  // layer 2 (reuse hq buffers) + residual add in epilogue
  k_bnstats<<<C, 256, 0, stream>>>(c1, gamma2, beta2, scale2, shift2);
  k_quant<<<NPIX * C / 256, 256, 0, stream>>>(c1, scale2, shift2, hq);
  k_conv<true><<<NPIX / 64, 256, 0, stream>>>(hq, weff + (size_t)C * KTOT, x, out);
}
